// Decoders_40140764349027
// MI455X (gfx1250) — compile-verified, matrix operand order checked by reading, not run on hardware
//
#include <hip/hip_runtime.h>

typedef __attribute__((ext_vector_type(16))) _Float16 v16h;
typedef __attribute__((ext_vector_type(8)))  _Float16 v8h;
typedef __attribute__((ext_vector_type(8)))  float    v8f;

#define NPTS 262144
#define TSZ  (1u << 19)

// Fragment ids inside the packed-weight arena (1 frag = 32 lanes * 16 halves = 1024 B)
#define FR_FW0   0
#define FR_FW1   1
#define FR_FW2   2
#define FR_FW3   3
#define FR_OSDF  4
#define FR_ORGB0 5   // 3 frags (K = 80 padded to 96)
#define FR_SW0   8   // 16 frags (nt 0..15, K=32)
#define FR_SW1   24  // 128 frags (nt*8 + kc)
#define FR_SW2   152 // 128 frags
#define FR_SWO   280 // 16 frags (nt 0..1, kc 0..7), O=25 padded to 32
#define N_FRAGS  296

// ---------------------------------------------------------------- WMMA helpers

__device__ __forceinline__ v8f wmma_f16(v16h a, v16h b, v8f c) {
  return __builtin_amdgcn_wmma_f32_16x16x32_f16(false, a, false, b, (short)0, c,
                                                false, false);
}

// B fragment: one 32B load per lane (pre-swizzled by pack_weights)
__device__ __forceinline__ v16h load_bfrag(const _Float16* frags, int id, int lane) {
  return *(const v16h*)(frags + (size_t)id * 512 + lane * 16);
}

// A fragment from an activation tile act[16][rs] (f16): row M = lane&15,
// halves 0..7 -> K = kc*32 + (lane>=16?8:0) + j ; halves 8..15 -> +16.
__device__ __forceinline__ v16h load_afrag(const _Float16* act, int rs, int lane, int kc) {
  const int m  = lane & 15;
  const int kb = kc * 32 + ((lane >> 4) << 3);
  const _Float16* r = act + m * rs + kb;
  v8h lo = *(const v8h*)(r);
  v8h hi = *(const v8h*)(r + 16);
  v16h a;
#pragma unroll
  for (int j = 0; j < 8; ++j) { a[j] = lo[j]; a[j + 8] = hi[j]; }
  return a;
}

// C tile (f32 accum: VGPR v -> row M = v + (lane>=16?8:0), col N = lane&15)
// -> bias (+ReLU) -> f16 back into the activation tile at column colBase+N.
__device__ __forceinline__ void store_c16(_Float16* act, int rs, int lane, int colBase,
                                          v8f c, float bias, bool relu) {
  const int n  = colBase + (lane & 15);
  const int rb = (lane >> 4) * 8;
#pragma unroll
  for (int v = 0; v < 8; ++v) {
    float x = c[v] + bias;
    if (relu) x = fmaxf(x, 0.0f);
    act[(rb + v) * rs + n] = (_Float16)x;
  }
}

// ------------------------------------------------------------ weight packing

__global__ void pack_weights_kernel(const float* __restrict__ f_w0, const float* __restrict__ f_w1,
                                    const float* __restrict__ f_w2, const float* __restrict__ f_w3,
                                    const float* __restrict__ osdf_w, const float* __restrict__ orgb_w0,
                                    const float* __restrict__ s_w0, const float* __restrict__ s_w1,
                                    const float* __restrict__ s_w2, const float* __restrict__ s_wo,
                                    _Float16* __restrict__ frags) {
  const int id   = blockIdx.x;
  const int lane = threadIdx.x;
  const float* W = f_w0; int O = 16, I = 32, kc = 0, nt = 0; bool osdf = false;
  if (id == 0)            { W = f_w0;  O = 16;  I = 32; }
  else if (id <= 3)       { W = (id == 1 ? f_w1 : (id == 2 ? f_w2 : f_w3)); O = 16; I = 16; }
  else if (id == 4)       { W = osdf_w; O = 16; I = 16; osdf = true; }   // rows 1..16
  else if (id <= 7)       { W = orgb_w0; O = 16; I = 80; kc = id - 5; }
  else if (id <= 23)      { W = s_w0;  O = 256; I = 32;  nt = id - 8; }
  else if (id <= 151)     { int t = id - 24;  W = s_w1; O = 256; I = 256; nt = t >> 3; kc = t & 7; }
  else if (id <= 279)     { int t = id - 152; W = s_w2; O = 256; I = 256; nt = t >> 3; kc = t & 7; }
  else                    { int t = id - 280; W = s_wo; O = 25;  I = 256; nt = t >> 3; kc = t & 7; }

  const int n  = nt * 16 + (lane & 15);
  const int kb = kc * 32 + ((lane >> 4) << 3);
  _Float16* o = frags + (size_t)id * 512 + lane * 16;
#pragma unroll
  for (int j = 0; j < 16; ++j) {
    const int k = kb + (j < 8 ? j : j + 8);
    float v = 0.0f;
    if (k < I && n < O) v = osdf ? W[(n + 1) * 16 + k] : W[(size_t)n * I + k];
    o[j] = (_Float16)v;
  }
}

// ------------------------------------------------------------- hash encoding

__global__ __launch_bounds__(256) void encode_kernel(
    const float* __restrict__ p, const float* __restrict__ bound,
    const float* __restrict__ t_sdf, const float* __restrict__ t_color,
    const float* __restrict__ t_sem,
    _Float16* __restrict__ feat16, _Float16* __restrict__ cfeat16,
    _Float16* __restrict__ sfeat16, float* __restrict__ out) {
  const int pt = blockIdx.x * blockDim.x + threadIdx.x;
  if (pt >= NPTS) return;

  const float RES_TBL[16] = {16.f, 22.f, 30.f, 42.f, 58.f, 80.f, 111.f, 153.f,
                             212.f, 294.f, 406.f, 561.f, 776.f, 1072.f, 1482.f, 2048.f};
  const float px = p[pt * 3 + 0], py = p[pt * 3 + 1], pz = p[pt * 3 + 2];
  const float pnx = (px - bound[0]) / (bound[1] - bound[0]);
  const float pny = (py - bound[2]) / (bound[3] - bound[2]);
  const float pnz = (pz - bound[4]) / (bound[5] - bound[4]);

#pragma unroll 1
  for (int tbl = 0; tbl < 3; ++tbl) {
    const float* tab = (tbl == 0) ? t_sdf : (tbl == 1 ? t_color : t_sem);
    _Float16* fout   = (tbl == 0) ? feat16 : (tbl == 1 ? cfeat16 : sfeat16);
    const int colB   = (tbl == 0) ? 16 : (tbl == 1 ? 0 : 32);   // plane_feat column base

    alignas(16) _Float16 hb[32];
    alignas(16) float    pf[16];
#pragma unroll
    for (int lvl = 0; lvl < 16; ++lvl) {
      const float res = RES_TBL[lvl];
      const float sx = pnx * res, sy = pny * res, sz = pnz * res;
      const float fx = floorf(sx), fy = floorf(sy), fz = floorf(sz);
      const float rx = sx - fx, ry = sy - fy, rz = sz - fz;
      const unsigned ix = (unsigned)fx, iy = (unsigned)fy, iz = (unsigned)fz;
      const unsigned hy0 = iy * 2654435761u, hy1 = (iy + 1u) * 2654435761u;
      const unsigned hz0 = iz * 805459861u,  hz1 = (iz + 1u) * 805459861u;
      const float* lb = tab + (size_t)lvl * TSZ * 2;
      float f0 = 0.f, f1 = 0.f;
#pragma unroll
      for (int cor = 0; cor < 8; ++cor) {
        const unsigned hx  = ix + (unsigned)(cor >> 2);
        const unsigned idx = (hx ^ ((cor & 2) ? hy1 : hy0) ^ ((cor & 1) ? hz1 : hz0)) & (TSZ - 1u);
        const float2 e = *(const float2*)(lb + (size_t)idx * 2);
        const float w = ((cor & 4) ? rx : 1.f - rx) * ((cor & 2) ? ry : 1.f - ry) *
                        ((cor & 1) ? rz : 1.f - rz);
        f0 += w * e.x; f1 += w * e.y;
      }
      hb[lvl * 2] = (_Float16)f0; hb[lvl * 2 + 1] = (_Float16)f1;
      if (lvl >= 8) { pf[(lvl - 8) * 2] = f0; pf[(lvl - 8) * 2 + 1] = f1; }
    }
    // f16 feature row (64 B, vectorized)
    _Float16* fo = fout + (size_t)pt * 32;
#pragma unroll
    for (int q = 0; q < 4; ++q) ((uint4*)fo)[q] = ((const uint4*)hb)[q];
    // plane_feat slice (f32, levels 8..15)
    float* po = out + (size_t)NPTS * 29 + (size_t)pt * 48 + colB;
#pragma unroll
    for (int q = 0; q < 4; ++q) ((float4*)po)[q] = ((const float4*)pf)[q];
  }
}

// ------------------------------------------------- geometry (f MLP) + rgb head

__global__ __launch_bounds__(256) void geo_rgb_kernel(
    const _Float16* __restrict__ feat16, const _Float16* __restrict__ cfeat16,
    const _Float16* __restrict__ sfeat16, const _Float16* __restrict__ frags,
    const float* __restrict__ f_b0, const float* __restrict__ f_b1,
    const float* __restrict__ f_b2, const float* __restrict__ f_b3,
    const float* __restrict__ osdf_w, const float* __restrict__ osdf_b,
    const float* __restrict__ orgb_b0, const float* __restrict__ orgb_w1,
    const float* __restrict__ orgb_b1, float* __restrict__ out) {
  __shared__ _Float16 actLds[8][16][96];
  const int lane = threadIdx.x & 31;
  const int wave = threadIdx.x >> 5;
  _Float16* act  = &actLds[wave][0][0];
  const int m0 = (blockIdx.x * 8 + wave) * 16;
  const int m  = lane & 15;
  const int hi = lane >> 4;

  // stage feat tile (16 pts x 32) -> act cols 0..31
  {
    const _Float16* src = feat16 + ((size_t)(m0 + m)) * 32 + hi * 16;
    *(v8h*)(act + m * 96 + hi * 16)     = *(const v8h*)(src);
    *(v8h*)(act + m * 96 + hi * 16 + 8) = *(const v8h*)(src + 8);
  }
  // f0: 32 -> 16
  { v16h a = load_afrag(act, 96, lane, 0); v8f c = {};
    c = wmma_f16(a, load_bfrag(frags, FR_FW0, lane), c);
    store_c16(act, 96, lane, 0, c, f_b0[m], true); }
  // f1..f3: 16 -> 16 (K zero-padded in the B fragments)
  { v16h a = load_afrag(act, 96, lane, 0); v8f c = {};
    c = wmma_f16(a, load_bfrag(frags, FR_FW1, lane), c);
    store_c16(act, 96, lane, 0, c, f_b1[m], true); }
  { v16h a = load_afrag(act, 96, lane, 0); v8f c = {};
    c = wmma_f16(a, load_bfrag(frags, FR_FW2, lane), c);
    store_c16(act, 96, lane, 0, c, f_b2[m], true); }
  { v16h a = load_afrag(act, 96, lane, 0); v8f c = {};
    c = wmma_f16(a, load_bfrag(frags, FR_FW3, lane), c);
    store_c16(act, 96, lane, 0, c, f_b3[m], true); }

  // sdf scalar (osdf row 0) from h3, then sdf_feat (osdf rows 1..16) via WMMA
  v16h aOs = load_afrag(act, 96, lane, 0);
  if (lane < 16) {
    float s = osdf_b[0];
#pragma unroll
    for (int k = 0; k < 16; ++k) s += (float)act[lane * 96 + k] * osdf_w[k];
    out[(size_t)(m0 + lane) * 29 + 3] = tanhf(s);
  }
  { v8f c = {};
    c = wmma_f16(aOs, load_bfrag(frags, FR_OSDF, lane), c);
    store_c16(act, 96, lane, 0, c, osdf_b[1 + m], false); }

  // assemble rgb input: [sdf_feat(0..15) | c_feat(16..47) | s_feat(48..79)]
  {
    const _Float16* cs = cfeat16 + ((size_t)(m0 + m)) * 32 + hi * 16;
    *(v8h*)(act + m * 96 + 16 + hi * 16)     = *(const v8h*)(cs);
    *(v8h*)(act + m * 96 + 16 + hi * 16 + 8) = *(const v8h*)(cs + 8);
    const _Float16* ss = sfeat16 + ((size_t)(m0 + m)) * 32 + hi * 16;
    *(v8h*)(act + m * 96 + 48 + hi * 16)     = *(const v8h*)(ss);
    *(v8h*)(act + m * 96 + 48 + hi * 16 + 8) = *(const v8h*)(ss + 8);
  }
  // rgb hidden: 80(->96) -> 16, three accumulating K-chunks
  {
    v8f c = {};
#pragma unroll
    for (int kc = 0; kc < 3; ++kc) {
      v16h a = load_afrag(act, 96, lane, kc);
      c = wmma_f16(a, load_bfrag(frags, FR_ORGB0 + kc, lane), c);
    }
    store_c16(act, 96, lane, 0, c, orgb_b0[m], true);
  }
  // rgb out: 16 -> 3 + sigmoid (VALU, one point per lane 0..15)
  if (lane < 16) {
    const size_t pt = (size_t)(m0 + lane);
#pragma unroll
    for (int o = 0; o < 3; ++o) {
      float s = orgb_b1[o];
#pragma unroll
      for (int k = 0; k < 16; ++k) s += (float)act[lane * 96 + k] * orgb_w1[o * 16 + k];
      out[pt * 29 + o] = 1.0f / (1.0f + expf(-s));
    }
  }
}

// ------------------------------------------------------------- semantic MLP

__global__ __launch_bounds__(128) void semantic_kernel(
    const _Float16* __restrict__ sfeat16, const _Float16* __restrict__ frags,
    const float* __restrict__ s_b0, const float* __restrict__ s_b1,
    const float* __restrict__ s_b2, const float* __restrict__ s_bo,
    float* __restrict__ out) {
  __shared__ _Float16 actLds[4][16][256];
  const int lane = threadIdx.x & 31;
  const int wave = threadIdx.x >> 5;
  _Float16* act  = &actLds[wave][0][0];
  const int m0 = (blockIdx.x * 4 + wave) * 16;
  const int m  = lane & 15;
  const int hi = lane >> 4;

  // stage s_feat tile -> act cols 0..31
  {
    const _Float16* src = sfeat16 + ((size_t)(m0 + m)) * 32 + hi * 16;
    *(v8h*)(act + m * 256 + hi * 16)     = *(const v8h*)(src);
    *(v8h*)(act + m * 256 + hi * 16 + 8) = *(const v8h*)(src + 8);
  }
  // L0: 32 -> 256
  {
    v16h a0 = load_afrag(act, 256, lane, 0);
#pragma unroll 1
    for (int nt = 0; nt < 16; ++nt) {
      v8f c = {};
      c = wmma_f16(a0, load_bfrag(frags, FR_SW0 + nt, lane), c);
      store_c16(act, 256, lane, nt * 16, c, s_b0[nt * 16 + m], true);
    }
  }
  // L1: 256 -> 256 (hoist all A fragments, then overwrite act with outputs)
  {
    v16h aF[8];
#pragma unroll
    for (int kc = 0; kc < 8; ++kc) aF[kc] = load_afrag(act, 256, lane, kc);
#pragma unroll 1
    for (int nt = 0; nt < 16; ++nt) {
      v8f c = {};
#pragma unroll
      for (int kc = 0; kc < 8; ++kc)
        c = wmma_f16(aF[kc], load_bfrag(frags, FR_SW1 + nt * 8 + kc, lane), c);
      store_c16(act, 256, lane, nt * 16, c, s_b1[nt * 16 + m], true);
    }
  }
  // L2: 256 -> 256
  {
    v16h aF[8];
#pragma unroll
    for (int kc = 0; kc < 8; ++kc) aF[kc] = load_afrag(act, 256, lane, kc);
#pragma unroll 1
    for (int nt = 0; nt < 16; ++nt) {
      v8f c = {};
#pragma unroll
      for (int kc = 0; kc < 8; ++kc)
        c = wmma_f16(aF[kc], load_bfrag(frags, FR_SW2 + nt * 8 + kc, lane), c);
      store_c16(act, 256, lane, nt * 16, c, s_b2[nt * 16 + m], true);
    }
  }
  // Out: 256 -> 25 (padded to 32), write raw[:, 4:29]
  {
    v16h aF[8];
#pragma unroll
    for (int kc = 0; kc < 8; ++kc) aF[kc] = load_afrag(act, 256, lane, kc);
#pragma unroll 1
    for (int nt = 0; nt < 2; ++nt) {
      v8f c = {};
#pragma unroll
      for (int kc = 0; kc < 8; ++kc)
        c = wmma_f16(aF[kc], load_bfrag(frags, FR_SWO + nt * 8 + kc, lane), c);
      const int n = nt * 16 + m;
      if (n < 25) {
        const float b = s_bo[n];
#pragma unroll
        for (int v = 0; v < 8; ++v) {
          const size_t pt = (size_t)(m0 + hi * 8 + v);
          out[pt * 29 + 4 + n] = c[v] + b;
        }
      }
    }
  }
}

// ----------------------------------------------------------------- launcher

extern "C" void kernel_launch(void* const* d_in, const int* in_sizes, int n_in,
                              void* d_out, int out_size, void* d_ws, size_t ws_size,
                              hipStream_t stream) {
  (void)in_sizes; (void)n_in; (void)out_size; (void)ws_size;
  const float* p       = (const float*)d_in[0];
  const float* bound   = (const float*)d_in[1];
  const float* t_sdf   = (const float*)d_in[2];
  const float* t_color = (const float*)d_in[3];
  const float* t_sem   = (const float*)d_in[4];
  const float* f_w0 = (const float*)d_in[5];  const float* f_b0 = (const float*)d_in[6];
  const float* f_w1 = (const float*)d_in[7];  const float* f_b1 = (const float*)d_in[8];
  const float* f_w2 = (const float*)d_in[9];  const float* f_b2 = (const float*)d_in[10];
  const float* f_w3 = (const float*)d_in[11]; const float* f_b3 = (const float*)d_in[12];
  const float* osdf_w  = (const float*)d_in[13]; const float* osdf_b  = (const float*)d_in[14];
  const float* orgb_w0 = (const float*)d_in[15]; const float* orgb_b0 = (const float*)d_in[16];
  const float* orgb_w1 = (const float*)d_in[17]; const float* orgb_b1 = (const float*)d_in[18];
  const float* s_w0 = (const float*)d_in[19]; const float* s_b0 = (const float*)d_in[20];
  const float* s_w1 = (const float*)d_in[21]; const float* s_b1 = (const float*)d_in[22];
  const float* s_w2 = (const float*)d_in[23]; const float* s_b2 = (const float*)d_in[24];
  const float* s_wo = (const float*)d_in[25]; const float* s_bo = (const float*)d_in[26];

  char* ws = (char*)d_ws;
  _Float16* feat16  = (_Float16*)(ws);
  _Float16* cfeat16 = (_Float16*)(ws + (size_t)NPTS * 64);
  _Float16* sfeat16 = (_Float16*)(ws + (size_t)NPTS * 128);
  _Float16* frags   = (_Float16*)(ws + (size_t)NPTS * 192);
  float* out = (float*)d_out;

  pack_weights_kernel<<<N_FRAGS, 32, 0, stream>>>(f_w0, f_w1, f_w2, f_w3, osdf_w,
                                                  orgb_w0, s_w0, s_w1, s_w2, s_wo, frags);
  encode_kernel<<<NPTS / 256, 256, 0, stream>>>(p, bound, t_sdf, t_color, t_sem,
                                                feat16, cfeat16, sfeat16, out);
  geo_rgb_kernel<<<NPTS / 128, 256, 0, stream>>>(feat16, cfeat16, sfeat16, frags,
                                                 f_b0, f_b1, f_b2, f_b3, osdf_w, osdf_b,
                                                 orgb_b0, orgb_w1, orgb_b1, out);
  semantic_kernel<<<NPTS / 64, 128, 0, stream>>>(sfeat16, frags, s_b0, s_b1, s_b2,
                                                 s_bo, out);
}
